// Model21_80573586473436
// MI455X (gfx1250) — matrix operand order repaired and checked by reading, not compiled, not run on hardware
//
#include <hip/hip_runtime.h>
#include <hip/hip_fp16.h>
#include <math.h>

// ---------------------------------------------------------------------------
// MI455X (gfx1250) implementation of the PointMLP-style reference.
// All conv1/linear layers run through a WMMA f16->f32 GEMM with fused
// BN/bias/residual/GELU epilogue. Activations are fp16 end-to-end.
// GEMM: block 128 threads = 4 waves; block tile 128(M) x 32(N), K-step 32;
// each wave owns a 32x32 output tile = 4 v_wmma_f32_16x16x32_f16 per K-step
// (A/B fragments each reused twice -> 2 ds_load_b128 per WMMA).
// ---------------------------------------------------------------------------

typedef __attribute__((ext_vector_type(16))) _Float16 v16h;
typedef __attribute__((ext_vector_type(8)))  _Float16 v8h;
typedef __attribute__((ext_vector_type(8)))  float    v8f;

#define RSQ_1PEPS 0.9999950000374997f   // 1/sqrt(1 + 1e-5)

__device__ __forceinline__ float gelu_f(float x) {
    return 0.5f * x * (1.0f + erff(x * 0.70710678118654752f));
}

// ------------------------- elementwise helpers -----------------------------

__global__ void k_f2h(const float* __restrict__ s, _Float16* __restrict__ d, int n) {
    int i = blockIdx.x * blockDim.x + threadIdx.x;
    if (i < n) d[i] = (_Float16)s[i];
}

// x: [B][3][N] f32 -> xyz: [B][N][3] f32
__global__ void k_build_xyz(const float* __restrict__ x, float* __restrict__ xyz,
                            int B, int N) {
    int e = blockIdx.x * blockDim.x + threadIdx.x;
    if (e >= B * N) return;
    int b = e / N, n = e % N;
    const float* xb = x + (size_t)b * 3 * N;
    float* o = xyz + (size_t)e * 3;
    o[0] = xb[n]; o[1] = xb[N + n]; o[2] = xb[2 * N + n];
}

// furthest point sampling, one block (256 threads) per batch. Matches the
// reference: start index 0, dist init 1e10, argmax picks first max on ties.
__global__ void __launch_bounds__(256) k_fps(const float* __restrict__ xyz,
                                             int N, int S, int* __restrict__ fi) {
    __shared__ float dist[1024];
    __shared__ float rv[256];
    __shared__ int   ri[256];
    __shared__ int   far_s;
    const int tid = threadIdx.x;
    const int b = blockIdx.x;
    const float* xb = xyz + (size_t)b * N * 3;
    for (int j = tid; j < N; j += 256) dist[j] = 1e10f;
    if (tid == 0) far_s = 0;
    __syncthreads();
    for (int i = 0; i < S; ++i) {
        int far = far_s;
        if (tid == 0) fi[b * S + i] = far;
        float cx = xb[far * 3 + 0], cy = xb[far * 3 + 1], cz = xb[far * 3 + 2];
        float bv = -1.0f; int bi = 0;
        for (int j = tid; j < N; j += 256) {
            float dx = xb[j * 3 + 0] - cx;
            float dy = xb[j * 3 + 1] - cy;
            float dz = xb[j * 3 + 2] - cz;
            float d = dx * dx + dy * dy + dz * dz;
            float nd = fminf(dist[j], d);
            dist[j] = nd;
            if (nd > bv) { bv = nd; bi = j; }
        }
        rv[tid] = bv; ri[tid] = bi;
        __syncthreads();
        for (int s2 = 128; s2 > 0; s2 >>= 1) {
            if (tid < s2) {
                if (rv[tid + s2] > rv[tid] ||
                    (rv[tid + s2] == rv[tid] && ri[tid + s2] < ri[tid])) {
                    rv[tid] = rv[tid + s2]; ri[tid] = ri[tid + s2];
                }
            }
            __syncthreads();
        }
        if (tid == 0) far_s = ri[0];
        __syncthreads();
    }
}

__global__ void k_gather_xyz(const float* __restrict__ xyz, const int* __restrict__ fi,
                             float* __restrict__ nxyz, int B, int N, int S) {
    int e = blockIdx.x * blockDim.x + threadIdx.x;
    if (e >= B * S) return;
    int b = e / S;
    int j = fi[e];
    const float* src = xyz + ((size_t)b * N + j) * 3;
    float* dst = nxyz + (size_t)e * 3;
    dst[0] = src[0]; dst[1] = src[1]; dst[2] = src[2];
}

// one wave (32 threads) per query point; extract 32 nearest in ascending
// order (ties -> lower index) to match lax.top_k(-d, k).
__global__ void k_knn(const float* __restrict__ xyz, const float* __restrict__ nxyz,
                      int N, int S, int* __restrict__ idx) {
    extern __shared__ float d[];
    const int lane = threadIdx.x;
    const int bs = blockIdx.x;
    const int b = bs / S;
    const float* xb = xyz + (size_t)b * N * 3;
    float qx = nxyz[(size_t)bs * 3 + 0];
    float qy = nxyz[(size_t)bs * 3 + 1];
    float qz = nxyz[(size_t)bs * 3 + 2];
    for (int j = lane; j < N; j += 32) {
        float dx = xb[j * 3 + 0] - qx;
        float dy = xb[j * 3 + 1] - qy;
        float dz = xb[j * 3 + 2] - qz;
        d[j] = dx * dx + dy * dy + dz * dz;
    }
    for (int k = 0; k < 32; ++k) {
        float bv = 3.0e38f; int bi = 0x7fffffff;
        for (int j = lane; j < N; j += 32)
            if (d[j] < bv || (d[j] == bv && j < bi)) { bv = d[j]; bi = j; }
        for (int off = 16; off > 0; off >>= 1) {
            float ov = __shfl_xor(bv, off);
            int   oi = __shfl_xor(bi, off);
            if (ov < bv || (ov == bv && oi < bi)) { bv = ov; bi = oi; }
        }
        if (lane == 0) { idx[(size_t)bs * 32 + k] = bi; d[bi] = 3.0e38f; }
    }
}

// build grouped input [CH][c][32] fp16: rows 0..d-1 = grouped - center,
// rows d..2d-1 = center broadcast.  h: [B][d][Nin] fp16.
__global__ void k_group(const _Float16* __restrict__ h, const int* __restrict__ fi,
                        const int* __restrict__ idxk, _Float16* __restrict__ X,
                        int Nin, int S, int c, int bg0, int CH) {
    size_t e = (size_t)blockIdx.x * blockDim.x + threadIdx.x;
    size_t total = (size_t)CH * c * 32;
    if (e >= total) return;
    int kk = (int)(e & 31);
    int ch = (int)((e >> 5) % c);
    int li = (int)(e / ((size_t)c * 32));
    int bg = bg0 + li;
    int b = bg / S, s = bg % S;
    int dh = c >> 1;
    int center = fi[b * S + s];
    const _Float16* hb = h + (size_t)b * dh * Nin;
    float v;
    if (ch < dh) {
        int j = idxk[((size_t)b * S + s) * 32 + kk];
        v = (float)hb[(size_t)ch * Nin + j] - (float)hb[(size_t)ch * Nin + center];
    } else {
        v = (float)hb[(size_t)(ch - dh) * Nin + center];
    }
    X[e] = (_Float16)v;
}

// X: [CH][c][32] -> hn: [B][c][S] (max over k)
__global__ void k_maxpool(const _Float16* __restrict__ X, _Float16* __restrict__ hn,
                          int c, int S, int bg0, int CH) {
    int e = blockIdx.x * blockDim.x + threadIdx.x;
    if (e >= CH * c) return;
    int ch = e % c;
    int li = e / c;
    int bg = bg0 + li;
    int b = bg / S, g = bg % S;
    const _Float16* xr = X + (size_t)li * c * 32 + (size_t)ch * 32;
    float m = -3.0e38f;
#pragma unroll
    for (int k = 0; k < 32; ++k) m = fmaxf(m, (float)xr[k]);
    hn[((size_t)b * c + ch) * S + g] = (_Float16)m;
}

// h: [16][1024][64] -> featT: [2048][16]  (rows 0..1023 max, 1024..2047 mean)
__global__ void k_featpool(const _Float16* __restrict__ h, _Float16* __restrict__ featT) {
    int e = blockIdx.x * blockDim.x + threadIdx.x;
    if (e >= 16 * 1024) return;
    int b = e >> 10, ch = e & 1023;
    const _Float16* r = h + ((size_t)b * 1024 + ch) * 64;
    float mx = -3.0e38f, sm = 0.f;
    for (int k = 0; k < 64; ++k) { float v = (float)r[k]; mx = fmaxf(mx, v); sm += v; }
    featT[ch * 16 + b] = (_Float16)mx;
    featT[(1024 + ch) * 16 + b] = (_Float16)(sm * (1.0f / 64.0f));
}

// fused softmax(QK^T/sqrt(d)) @ V, one wave per (batch, head, row).
// qkv: [batch][384][n] fp16 (q rows 0..127, k 128..255, v 256..383)
// out: [batch][128][n] fp16 (row = head*32 + d)
__global__ void __launch_bounds__(256) k_attn(const _Float16* __restrict__ qkv,
                                              _Float16* __restrict__ out,
                                              int n, int totalRows) {
    extern __shared__ float sm[];
    const int lane = threadIdx.x & 31;
    const int w = threadIdx.x >> 5;
    const int row = blockIdx.x * 8 + w;
    const bool valid = row < totalRows;
    const int rr = valid ? row : 0;
    const int b = rr / (4 * n);
    const int h = (rr / n) & 3;
    const int i = rr % n;
    float* pr = sm + (size_t)w * n;
    const _Float16* qb = qkv + (size_t)b * 384 * n + (size_t)(h * 32) * n;
    const _Float16* kb = qb + (size_t)128 * n;
    const _Float16* vb = qb + (size_t)256 * n;
    float qv[32];
#pragma unroll
    for (int dd = 0; dd < 32; ++dd) qv[dd] = (float)qb[(size_t)dd * n + i];
    for (int j = lane; j < n; j += 32) {
        float s = 0.f;
#pragma unroll
        for (int dd = 0; dd < 32; ++dd) s += qv[dd] * (float)kb[(size_t)dd * n + j];
        pr[j] = s * 0.17677669529663689f;   // 1/sqrt(32)
    }
    float mx = -3.0e38f;
    for (int j = lane; j < n; j += 32) mx = fmaxf(mx, pr[j]);
    for (int off = 16; off > 0; off >>= 1) mx = fmaxf(mx, __shfl_xor(mx, off));
    float sum = 0.f;
    for (int j = lane; j < n; j += 32) { float e = __expf(pr[j] - mx); pr[j] = e; sum += e; }
    for (int off = 16; off > 0; off >>= 1) sum += __shfl_xor(sum, off);
    float inv = 1.0f / sum;
    __syncthreads();  // uniform; make pr[] visible across lanes
    float acc = 0.f;
    for (int j = 0; j < n; ++j) acc += pr[j] * (float)vb[(size_t)lane * n + j];
    if (valid)
        out[(size_t)b * 128 * n + (size_t)(h * 32 + lane) * n + i] = (_Float16)(acc * inv);
}

// ------------------------------- WMMA GEMM ---------------------------------
// D[z][m][n] = epi( sum_k A[m][k]*B[z][k][n] ), A fp16 [M][K] shared across
// the batch, B fp16 [batch][K][N].  Block tile 128(M) x 32(N), K-step 32.
// Wave w owns rows [w*32, w*32+32): 4 WMMA accumulators (2 M-frags x 2 N-frags).
// epi: v=acc; +bias[m]; *gamma[m]*rsqrt(1+eps)+beta[m]; +res; gelu; ->fp16/fp32.

__device__ __forceinline__ v16h frag16(const _Float16* p, int off2) {
    v8h lo = *(const v8h*)(p);
    v8h hi = *(const v8h*)(p + off2);
    return __builtin_shufflevector(lo, hi, 0, 1, 2, 3, 4, 5, 6, 7,
                                   8, 9, 10, 11, 12, 13, 14, 15);
}

__global__ void __launch_bounds__(128) wmma_gemm(
    const _Float16* __restrict__ A, const _Float16* __restrict__ B,
    _Float16* __restrict__ out16, float* __restrict__ out32, int o32ld,
    const float* __restrict__ bias, const float* __restrict__ gamma,
    const float* __restrict__ beta, const _Float16* __restrict__ res,
    int M, int N, int K, int doGelu) {
    __shared__ __align__(16) _Float16 As[128 * 40];
    __shared__ __align__(16) _Float16 Bt[32 * 40];
    const int tid = threadIdx.x;
    const int lane = tid & 31;
    const int w = tid >> 5;
    const int n0 = blockIdx.x * 32;
    const int m0 = blockIdx.y * 128;
    const size_t bKN = (size_t)blockIdx.z * K * N;
    const size_t bMN = (size_t)blockIdx.z * M * N;
    const _Float16* Bb = B + bKN;
    v8f acc00 = {0.f, 0.f, 0.f, 0.f, 0.f, 0.f, 0.f, 0.f};
    v8f acc01 = acc00, acc10 = acc00, acc11 = acc00;
    const bool fastA = (K % 8) == 0;
    const bool fastB = (N % 8) == 0;

    for (int k0 = 0; k0 < K; k0 += 32) {
        __syncthreads();
        {   // stage A tile [128][32] -> As (row stride 40 halves); thread = row
            int m = m0 + tid;
            _Float16* dst = &As[tid * 40];
            if (m < M && fastA && (k0 + 32) <= K) {
                const v8h* src = (const v8h*)(A + (size_t)m * K + k0);
                *(v8h*)(dst +  0) = src[0];
                *(v8h*)(dst +  8) = src[1];
                *(v8h*)(dst + 16) = src[2];
                *(v8h*)(dst + 24) = src[3];
            } else {
                // branchless tail: clamped address + select-zero
                int ms = m < M ? m : (M - 1);
                const _Float16* src = A + (size_t)ms * K;
                for (int q = 0; q < 32; ++q) {
                    int k = k0 + q;
                    int ks = k < K ? k : (K - 1);
                    _Float16 v = src[ks];
                    dst[q] = (m < M && k < K) ? v : (_Float16)0.f;
                }
            }
        }
        {   // stage B tile [32K][32N] transposed -> Bt[n][k]
            int kk = tid >> 2, nn0 = (tid & 3) << 3;
            int k = k0 + kk;
            if (k < K && fastB && (n0 + nn0 + 8) <= N) {
                v8h v = *(const v8h*)(Bb + (size_t)k * N + n0 + nn0);
#pragma unroll
                for (int q = 0; q < 8; ++q) Bt[(nn0 + q) * 40 + kk] = v[q];
            } else {
                int ks = k < K ? k : (K - 1);
                const _Float16* src = Bb + (size_t)ks * N;
                for (int q = 0; q < 8; ++q) {
                    int n = n0 + nn0 + q;
                    int ns = n < N ? n : (N - 1);
                    _Float16 v = src[ns];
                    Bt[(nn0 + q) * 40 + kk] = (k < K && n < N) ? v : (_Float16)0.f;
                }
            }
            if (k0 + 32 + kk < K)
                __builtin_prefetch(Bb + (size_t)(k0 + 32 + kk) * N + n0 + nn0, 0, 1);
        }
        __syncthreads();

        // fragment gathers per the gfx1250 16-bit WMMA VGPR layouts
        const int bc = lane & 15;
        const int koff = (lane >> 4) << 3;          // A half-offset: 0 or 8
        const int kb = (lane >> 4) << 3;            // B half-offset: 0 or 8
        const int ar0 = (w << 5) + bc;
        v16h a0 = frag16(&As[ar0 * 40 + koff], 16);
        v16h a1 = frag16(&As[(ar0 + 16) * 40 + koff], 16);
        v16h b0 = frag16(&Bt[bc * 40 + kb], 16);
        v16h b1 = frag16(&Bt[(bc + 16) * 40 + kb], 16);

        acc00 = __builtin_amdgcn_wmma_f32_16x16x32_f16(false, a0, false, b0,
                                                       (short)0, acc00, false, false);
        acc01 = __builtin_amdgcn_wmma_f32_16x16x32_f16(false, a0, false, b1,
                                                       (short)0, acc01, false, false);
        acc10 = __builtin_amdgcn_wmma_f32_16x16x32_f16(false, a1, false, b0,
                                                       (short)0, acc10, false, false);
        acc11 = __builtin_amdgcn_wmma_f32_16x16x32_f16(false, a1, false, b1,
                                                       (short)0, acc11, false, false);
    }

    // epilogue (C/D layout: VGPR r -> M = r + 8*(lane>>4), N = lane&15)
    const int nl = lane & 15;
    const int mrow = ((lane >> 4) << 3);
#pragma unroll
    for (int mi = 0; mi < 2; ++mi) {
#pragma unroll
        for (int f = 0; f < 2; ++f) {
            v8f acc = mi ? (f ? acc11 : acc10) : (f ? acc01 : acc00);
            int n = n0 + (f << 4) + nl;
            if (n >= N) continue;
            int mbase = m0 + (w << 5) + (mi << 4) + mrow;
#pragma unroll
            for (int r = 0; r < 8; ++r) {
                int m = mbase + r;
                if (m >= M) continue;
                float v = acc[r];
                if (bias)  v += bias[m];
                if (gamma) v = v * (gamma[m] * RSQ_1PEPS) + beta[m];
                if (res)   v += (float)res[bMN + (size_t)m * N + n];
                if (doGelu) v = gelu_f(v);
                if (out16) out16[bMN + (size_t)m * N + n] = (_Float16)v;
                if (out32) out32[(size_t)n * o32ld + m] = v;
            }
        }
    }
}

// ------------------------------- host driver -------------------------------

extern "C" void kernel_launch(void* const* d_in, const int* in_sizes, int n_in,
                              void* d_out, int out_size, void* d_ws, size_t ws_size,
                              hipStream_t stream) {
    (void)in_sizes; (void)n_in; (void)out_size; (void)ws_size;
    const int B = 16, N0 = 1024, HEADS = 4;
    const int S_LIST[4] = {512, 256, 128, 64};
    const int C_LIST[4] = {128, 256, 512, 1024};

    char* base = (char*)d_ws;
    size_t off = 0;
    auto alloc = [&](size_t bytes) -> void* {
        void* ptr = base + off;
        off = (off + bytes + 255) & ~(size_t)255;
        return ptr;
    };
    auto toh = [&](const void* src, size_t n) -> const _Float16* {
        _Float16* dst = (_Float16*)alloc(n * sizeof(_Float16));
        k_f2h<<<dim3((unsigned)((n + 255) / 256)), dim3(256), 0, stream>>>(
            (const float*)src, dst, (int)n);
        return dst;
    };

    int p = 0;
    const float* x = (const float*)d_in[p++];

    struct EmbP { const _Float16* Wh; const float *g, *b; int Cin, Cout; } emb[2];
    emb[0].Cin = 3;  emb[0].Cout = 64;
    emb[1].Cin = 64; emb[1].Cout = 64;
    for (int i = 0; i < 2; ++i) {
        const float* W = (const float*)d_in[p++];
        emb[i].g = (const float*)d_in[p++];
        emb[i].b = (const float*)d_in[p++];
        emb[i].Wh = toh(W, (size_t)emb[i].Cout * emb[i].Cin);
    }

    struct ResP { const _Float16 *W1h, *W2h; const float *g1, *b1, *g2, *b2; };
    struct TransP { const _Float16 *Wqkvh, *Wouth, *Wffnh;
                    const float *bout, *g, *b, *gf, *bf; };
    struct StageP { ResP pre[2]; TransP pret; ResP pos[2]; TransP post; } st[4];

    auto readRes = [&](int c) -> ResP {
        ResP r;
        const float* W1 = (const float*)d_in[p++];
        r.g1 = (const float*)d_in[p++]; r.b1 = (const float*)d_in[p++];
        const float* W2 = (const float*)d_in[p++];
        r.g2 = (const float*)d_in[p++]; r.b2 = (const float*)d_in[p++];
        r.W1h = toh(W1, (size_t)c * c);
        r.W2h = toh(W2, (size_t)c * c);
        return r;
    };
    auto readTrans = [&](int c) -> TransP {
        TransP t;
        const float* Wqkv = (const float*)d_in[p++];
        const float* Wout = (const float*)d_in[p++];
        t.bout = (const float*)d_in[p++];
        t.g = (const float*)d_in[p++]; t.b = (const float*)d_in[p++];
        const float* Wffn = (const float*)d_in[p++];
        t.gf = (const float*)d_in[p++]; t.bf = (const float*)d_in[p++];
        t.Wqkvh = toh(Wqkv, (size_t)384 * c);
        t.Wouth = toh(Wout, (size_t)c * 128);
        t.Wffnh = toh(Wffn, (size_t)c * c);
        return t;
    };
    for (int i = 0; i < 4; ++i) {
        int c = C_LIST[i];
        st[i].pre[0] = readRes(c); st[i].pre[1] = readRes(c);
        st[i].pret = readTrans(c);
        st[i].pos[0] = readRes(c); st[i].pos[1] = readRes(c);
        st[i].post = readTrans(c);
    }
    const float* cW1 = (const float*)d_in[p++]; const float* cb1 = (const float*)d_in[p++];
    const float* cg1 = (const float*)d_in[p++]; const float* cbt1 = (const float*)d_in[p++];
    const float* cW2 = (const float*)d_in[p++]; const float* cb2 = (const float*)d_in[p++];
    const float* cg2 = (const float*)d_in[p++]; const float* cbt2 = (const float*)d_in[p++];
    const float* cW3 = (const float*)d_in[p++]; const float* cb3 = (const float*)d_in[p++];
    const _Float16* cW1h = toh(cW1, (size_t)512 * 2048);
    const _Float16* cW2h = toh(cW2, (size_t)256 * 512);
    const _Float16* cW3h = toh(cW3, (size_t)40 * 256);

    // activation scratch
    const int CH = 512;                       // groups per chunk
    const size_t BIG = (size_t)CH * 1024 * 32;
    _Float16* bX   = (_Float16*)alloc(BIG * 2);
    _Float16* bT1  = (_Float16*)alloc(BIG * 2);
    _Float16* bT2  = (_Float16*)alloc(BIG * 2);
    _Float16* bQkv = (_Float16*)alloc((size_t)CH * 384 * 32 * 2);
    _Float16* bAO  = (_Float16*)alloc((size_t)CH * 128 * 32 * 2);
    _Float16* hA   = (_Float16*)alloc((size_t)1048576 * 2);
    _Float16* hB   = (_Float16*)alloc((size_t)1048576 * 2);
    _Float16* xh   = (_Float16*)alloc((size_t)16 * 3 * 1024 * 2);
    float* xyzA = (float*)alloc((size_t)16 * 1024 * 3 * 4);
    float* xyzB = (float*)alloc((size_t)16 * 1024 * 3 * 4);
    int* fi   = (int*)alloc((size_t)16 * 512 * 4);
    int* idxk = (int*)alloc((size_t)16 * 512 * 32 * 4);
    _Float16* featT = (_Float16*)alloc((size_t)2048 * 16 * 2);
    _Float16* z1T   = (_Float16*)alloc((size_t)512 * 16 * 2);
    _Float16* z2T   = (_Float16*)alloc((size_t)256 * 16 * 2);

    auto gemm = [&](const _Float16* A, const _Float16* Bm, _Float16* o16,
                    float* o32, int o32ld, const float* bias, const float* g,
                    const float* bt, const _Float16* res,
                    int M, int Nn, int Kk, int batch, int doGelu) {
        dim3 gr((Nn + 31) / 32, (M + 127) / 128, batch);
        wmma_gemm<<<gr, 128, 0, stream>>>(A, Bm, o16, o32, o32ld, bias, g, bt,
                                          res, M, Nn, Kk, doGelu);
    };

    k_f2h<<<(16 * 3 * 1024 + 255) / 256, 256, 0, stream>>>(x, xh, 16 * 3 * 1024);
    k_build_xyz<<<(B * N0 + 255) / 256, 256, 0, stream>>>(x, xyzA, B, N0);

    // embedding: [16,3,1024] -> [16,64,1024] -> [16,64,1024]
    gemm(emb[0].Wh, xh, hA, nullptr, 0, nullptr, emb[0].g, emb[0].b, nullptr,
         64, 1024, 3, B, 1);
    gemm(emb[1].Wh, hA, hB, nullptr, 0, nullptr, emb[1].g, emb[1].b, nullptr,
         64, 1024, 64, B, 1);

    _Float16* hcur = hB; _Float16* hnext = hA;
    float* xyzCur = xyzA; float* xyzNext = xyzB;
    int Nin = N0;

    for (int si = 0; si < 4; ++si) {
        int S = S_LIST[si], c = C_LIST[si];
        k_fps<<<B, 256, 0, stream>>>(xyzCur, Nin, S, fi);
        k_gather_xyz<<<(B * S + 255) / 256, 256, 0, stream>>>(xyzCur, fi, xyzNext,
                                                              B, Nin, S);
        k_knn<<<B * S, 32, (size_t)Nin * sizeof(float), stream>>>(xyzCur, xyzNext,
                                                                  Nin, S, idxk);
        int totalBG = B * S;
        for (int bg0 = 0; bg0 < totalBG; bg0 += CH) {
            int cur = (CH < totalBG - bg0) ? CH : (totalBG - bg0);
            size_t total = (size_t)cur * c * 32;
            k_group<<<(unsigned)((total + 255) / 256), 256, 0, stream>>>(
                hcur, fi, idxk, bX, Nin, S, c, bg0, cur);
            // pre resblocks
            gemm(st[si].pre[0].W1h, bX,  bT1, nullptr, 0, nullptr,
                 st[si].pre[0].g1, st[si].pre[0].b1, nullptr, c, 32, c, cur, 1);
            gemm(st[si].pre[0].W2h, bT1, bT2, nullptr, 0, nullptr,
                 st[si].pre[0].g2, st[si].pre[0].b2, bX,      c, 32, c, cur, 1);
            gemm(st[si].pre[1].W1h, bT2, bT1, nullptr, 0, nullptr,
                 st[si].pre[1].g1, st[si].pre[1].b1, nullptr, c, 32, c, cur, 1);
            gemm(st[si].pre[1].W2h, bT1, bX,  nullptr, 0, nullptr,
                 st[si].pre[1].g2, st[si].pre[1].b2, bT2,     c, 32, c, cur, 1);
            // pre transformer
            gemm(st[si].pret.Wqkvh, bX, bQkv, nullptr, 0, nullptr, nullptr,
                 nullptr, nullptr, 384, 32, c, cur, 0);
            int rows = cur * HEADS * 32;
            k_attn<<<(rows + 7) / 8, 256, (size_t)8 * 32 * sizeof(float), stream>>>(
                bQkv, bAO, 32, rows);
            gemm(st[si].pret.Wouth, bAO, bT1, nullptr, 0, st[si].pret.bout,
                 st[si].pret.g, st[si].pret.b, bX, c, 32, 128, cur, 1);
            gemm(st[si].pret.Wffnh, bT1, bT2, nullptr, 0, nullptr,
                 st[si].pret.gf, st[si].pret.bf, bT1, c, 32, c, cur, 1);
            k_maxpool<<<(cur * c + 255) / 256, 256, 0, stream>>>(bT2, hnext, c, S,
                                                                 bg0, cur);
        }
        // pos phase on h = hnext [B][c][S]
        _Float16* h = hnext;
        gemm(st[si].pos[0].W1h, h,   bT1, nullptr, 0, nullptr,
             st[si].pos[0].g1, st[si].pos[0].b1, nullptr, c, S, c, B, 1);
        gemm(st[si].pos[0].W2h, bT1, bT2, nullptr, 0, nullptr,
             st[si].pos[0].g2, st[si].pos[0].b2, h,       c, S, c, B, 1);
        gemm(st[si].pos[1].W1h, bT2, bT1, nullptr, 0, nullptr,
             st[si].pos[1].g1, st[si].pos[1].b1, nullptr, c, S, c, B, 1);
        gemm(st[si].pos[1].W2h, bT1, h,   nullptr, 0, nullptr,
             st[si].pos[1].g2, st[si].pos[1].b2, bT2,     c, S, c, B, 1);
        gemm(st[si].post.Wqkvh, h, bQkv, nullptr, 0, nullptr, nullptr, nullptr,
             nullptr, 384, S, c, B, 0);
        int rows = B * HEADS * S;
        k_attn<<<(rows + 7) / 8, 256, (size_t)8 * S * sizeof(float), stream>>>(
            bQkv, bAO, S, rows);
        gemm(st[si].post.Wouth, bAO, bT1, nullptr, 0, st[si].post.bout,
             st[si].post.g, st[si].post.b, h, c, S, 128, B, 1);
        gemm(st[si].post.Wffnh, bT1, h, nullptr, 0, nullptr,
             st[si].post.gf, st[si].post.bf, bT1, c, S, c, B, 1);
        // rotate buffers
        _Float16* th = hcur; hcur = hnext; hnext = th;
        float* tx = xyzCur; xyzCur = xyzNext; xyzNext = tx;
        Nin = S;
    }

    // classifier head
    k_featpool<<<(16 * 1024 + 255) / 256, 256, 0, stream>>>(hcur, featT);
    gemm(cW1h, featT, z1T, nullptr, 0, cb1, cg1, cbt1, nullptr, 512, 16, 2048, 1, 1);
    gemm(cW2h, z1T,   z2T, nullptr, 0, cb2, cg2, cbt2, nullptr, 256, 16, 512,  1, 1);
    gemm(cW3h, z2T, nullptr, (float*)d_out, 40, cb3, nullptr, nullptr, nullptr,
         40, 16, 256, 1, 0);
}
